// PointerGeneratorDecoder_84378927497887
// MI455X (gfx1250) — matrix operand order repaired and checked by reading, not compiled, not run on hardware
//
#include <hip/hip_runtime.h>
#include <hip/hip_bf16.h>
#include <stdint.h>

#define B_    16
#define S_    400
#define H_    512
#define E_    256
#define V_    32000
#define T_    64
#define NSTEP 63
#define MAXOOV 50
#define EXT_  32050
#define KXH   1280      // [emb(256) | context(512) | h(512)]
#define NG    2048      // 4*H gates
#define KV    1024      // [h_new(512) | context(512)]

typedef __attribute__((ext_vector_type(16))) __bf16 v16bf;
typedef __attribute__((ext_vector_type(8)))  float  v8f;
typedef __attribute__((ext_vector_type(4)))  unsigned u32x4;

union Frag { v16bf v; u32x4 q[2]; };

__device__ __forceinline__ unsigned short f2bf(float f) {
  union { float f; unsigned u; } v; v.f = f;
  unsigned r = v.u + 0x7FFFu + ((v.u >> 16) & 1u);   // round-to-nearest-even
  return (unsigned short)(r >> 16);
}
__device__ __forceinline__ float sigmoidf_(float x) { return 1.0f / (1.0f + __expf(-x)); }

// ---- WMMA fragment loaders (wave32 layouts, ISA 7.12.2) -------------------
// A: 16-bit 16x32 (MxK). lane: M = lane&15, half = lane>>4.
//    elements 0..7  -> K = k0 + 8*half + e         (contiguous 16B)
//    elements 8..15 -> K = k0 + 16 + 8*half + (e-8)(contiguous 16B)
__device__ __forceinline__ void loadA(Frag& a, const unsigned short* base, int ldk,
                                      int k0, int lane) {
  int m = lane & 15, half = lane >> 4;
  const unsigned short* r = base + (size_t)m * ldk + k0;
  a.q[0] = *reinterpret_cast<const u32x4*>(r + 8 * half);
  a.q[1] = *reinterpret_cast<const u32x4*>(r + 16 + 8 * half);
}
// B: 16-bit 32x16 (KxN), B[k][n] = W[n][k] with W row-major [N][ldk].
//    lane: N = n0 + (lane&15); elements 0..7 -> K = k0 + 8*half + e, 8..15 -> K = k0 + 16 + 8*half + (e-8)
__device__ __forceinline__ void loadB(Frag& b, const unsigned short* base, int ldk,
                                      int n0, int k0, int lane) {
  int n = lane & 15, half = lane >> 4;
  const unsigned short* r = base + (size_t)(n0 + n) * ldk + k0;
  b.q[0] = *reinterpret_cast<const u32x4*>(r + 8 * half);
  b.q[1] = *reinterpret_cast<const u32x4*>(r + 16 + 8 * half);
}

// ---- one-time preprocessing ------------------------------------------------
__global__ void k_cvt_wout(const float* __restrict__ W, unsigned short* __restrict__ o) {
  size_t n = (size_t)V_ * KV;
  for (size_t i = (size_t)blockIdx.x * blockDim.x + threadIdx.x; i < n;
       i += (size_t)gridDim.x * blockDim.x)
    o[i] = f2bf(W[i]);
}
__global__ void k_cvt_wcat(const float* __restrict__ Wih, const float* __restrict__ Whh,
                           unsigned short* __restrict__ o) {
  int n = NG * KXH;
  for (int i = blockIdx.x * blockDim.x + threadIdx.x; i < n; i += gridDim.x * blockDim.x) {
    int r = i / KXH, k = i % KXH;
    float v = (k < (E_ + H_)) ? Wih[r * (E_ + H_) + k] : Whh[r * H_ + (k - (E_ + H_))];
    o[i] = f2bf(v);
  }
}
__global__ void k_cvt_wa(const float* __restrict__ Wa, unsigned short* __restrict__ o) {
  int n = H_ * H_;
  for (int i = blockIdx.x * blockDim.x + threadIdx.x; i < n; i += gridDim.x * blockDim.x)
    o[i] = f2bf(Wa[i]);
}
__global__ void k_init_state(const float* __restrict__ hid, const float* __restrict__ cel,
                             float* __restrict__ h, float* __restrict__ c) {
  int i = blockIdx.x * blockDim.x + threadIdx.x;
  if (i < B_ * H_) { h[i] = hid[i]; c[i] = cel[i]; }
}

// ---- per-step phase 1: embedding lookup + proj = h @ Wa^T (bf16 WMMA) -----
__global__ void k_proj_embed(int t, const int* __restrict__ tgt,
                             const float* __restrict__ embedding,
                             const float* __restrict__ h,
                             unsigned short* __restrict__ h16,
                             unsigned short* __restrict__ xh16,
                             float* __restrict__ embf,
                             const unsigned short* __restrict__ wa16,
                             float* __restrict__ proj) {
  int tid = threadIdx.x;
  for (int i = tid; i < B_ * E_; i += blockDim.x) {        // embedding gather
    int b = i >> 8, e = i & 255;
    int y = tgt[b * T_ + t];
    float v = embedding[(size_t)y * E_ + e];
    embf[i] = v;
    xh16[b * KXH + e] = f2bf(v);
  }
  for (int i = tid; i < B_ * H_; i += blockDim.x) {        // h -> bf16 (A-matrix + xh)
    unsigned short hb = f2bf(h[i]);
    h16[i] = hb;
    int b = i >> 9, k = i & 511;
    xh16[b * KXH + (E_ + H_) + k] = hb;
  }
  __syncthreads();
  __threadfence_block();
  int w = tid >> 5, lane = tid & 31;
  int half = lane >> 4, n = lane & 15;
  for (int nt = w * 2; nt < w * 2 + 2; ++nt) {             // 16 waves x 2 tiles = 32 N-tiles
    int n0 = nt * 16;
    v8f acc = {};
    for (int kt = 0; kt < H_ / 32; ++kt) {
      Frag a, bfr;
      loadA(a, h16, H_, kt * 32, lane);
      loadB(bfr, wa16, H_, n0, kt * 32, lane);
      acc = __builtin_amdgcn_wmma_f32_16x16x32_bf16(false, a.v, false, bfr.v,
                                                    (short)0, acc, false, false);
    }
#pragma unroll
    for (int j = 0; j < 8; ++j)
      proj[(j + 8 * half) * H_ + n0 + n] = acc[j];
  }
}

// ---- phase 2: attention scores (masked) -----------------------------------
__global__ void k_scores(const float* __restrict__ proj, const float* __restrict__ enc,
                         const int* __restrict__ src_lens, float* __restrict__ scores) {
  int b = blockIdx.y;
  int w = threadIdx.x >> 5, lane = threadIdx.x & 31;
  int L = src_lens[b];
  const float* p = proj + b * H_;
  for (int sl = w; sl < 50; sl += 4) {
    int s = blockIdx.x * 50 + sl;
    const float* e = enc + ((size_t)b * S_ + s) * H_;
    float acc = 0.f;
    for (int k = lane; k < H_; k += 32) acc += p[k] * e[k];
    for (int m = 16; m; m >>= 1) acc += __shfl_xor(acc, m, 32);
    if (lane == 0) scores[b * S_ + s] = (s < L) ? acc : -1e9f;
  }
}

// ---- phase 3: softmax over S ----------------------------------------------
__global__ void k_softmax(const float* __restrict__ scores, float* __restrict__ attn) {
  __shared__ float red[256];
  int b = blockIdx.x, tid = threadIdx.x;
  float m = -1e30f;
  for (int s = tid; s < S_; s += 256) m = fmaxf(m, scores[b * S_ + s]);
  red[tid] = m; __syncthreads();
  for (int o = 128; o; o >>= 1) { if (tid < o) red[tid] = fmaxf(red[tid], red[tid + o]); __syncthreads(); }
  float mx = red[0]; __syncthreads();
  float sum = 0.f;
  for (int s = tid; s < S_; s += 256) sum += __expf(scores[b * S_ + s] - mx);
  red[tid] = sum; __syncthreads();
  for (int o = 128; o; o >>= 1) { if (tid < o) red[tid] += red[tid + o]; __syncthreads(); }
  float inv = 1.0f / red[0];
  for (int s = tid; s < S_; s += 256) attn[b * S_ + s] = __expf(scores[b * S_ + s] - mx) * inv;
}

// ---- phase 4: context = attn @ enc ----------------------------------------
__global__ void k_context(const float* __restrict__ attn, const float* __restrict__ enc,
                          float* __restrict__ ctx, unsigned short* __restrict__ xh16,
                          unsigned short* __restrict__ vin16) {
  int b = blockIdx.y;
  int hh = blockIdx.x * 64 + threadIdx.x;
  const float* a = attn + b * S_;
  const float* e = enc + (size_t)b * S_ * H_ + hh;
  float acc = 0.f;
  for (int s = 0; s < S_; ++s) acc += a[s] * e[(size_t)s * H_];
  ctx[b * H_ + hh] = acc;
  unsigned short bits = f2bf(acc);
  xh16[b * KXH + E_ + hh] = bits;   // x = [emb, context] portion of fused input
  vin16[b * KV + H_ + hh] = bits;   // vocab_input = [h_new, context]
}

// ---- phase 5: LSTM gates GEMM [16x1280]@[1280x2048] (bf16 WMMA) -----------
__global__ void k_gates(const unsigned short* __restrict__ xh16,
                        const unsigned short* __restrict__ wcat16,
                        const float* __restrict__ b_ih, const float* __restrict__ b_hh,
                        float* __restrict__ gates) {
  int w = threadIdx.x >> 5, lane = threadIdx.x & 31;
  int half = lane >> 4, n = lane & 15;
  int tile0 = blockIdx.x * 16 + w * 2;                     // 8 blocks x 8 waves x 2 = 128 tiles
  for (int nt = tile0; nt < tile0 + 2; ++nt) {
    int n0 = nt * 16;
    v8f acc = {};
    for (int kt = 0; kt < KXH / 32; ++kt) {
      Frag a, bfr;
      loadA(a, xh16, KXH, kt * 32, lane);
      loadB(bfr, wcat16, KXH, n0, kt * 32, lane);
      acc = __builtin_amdgcn_wmma_f32_16x16x32_bf16(false, a.v, false, bfr.v,
                                                    (short)0, acc, false, false);
    }
    int nn = n0 + n;
    float bias = b_ih[nn] + b_hh[nn];
#pragma unroll
    for (int j = 0; j < 8; ++j)
      gates[(j + 8 * half) * NG + nn] = acc[j] + bias;
  }
}

// ---- phase 6: LSTM pointwise + p_gen --------------------------------------
__global__ void k_lstm_pgen(const float* __restrict__ gates, float* __restrict__ h,
                            float* __restrict__ c, const float* __restrict__ ctx,
                            const float* __restrict__ embf,
                            unsigned short* __restrict__ vin16,
                            const float* __restrict__ W_pgen,
                            const float* __restrict__ b_pgen, float* __restrict__ pgen) {
  int tid = threadIdx.x;
  for (int i = tid; i < B_ * H_; i += blockDim.x) {
    int b = i >> 9, u = i & 511;
    const float* g = gates + b * NG;
    float ig = sigmoidf_(g[u]);
    float fg = sigmoidf_(g[H_ + u]);
    float gg = tanhf(g[2 * H_ + u]);
    float og = sigmoidf_(g[3 * H_ + u]);
    float cn = fg * c[i] + ig * gg;
    float hn = og * tanhf(cn);
    c[i] = cn; h[i] = hn;
    vin16[b * KV + u] = f2bf(hn);
  }
  __threadfence_block();
  __syncthreads();
  int w = tid >> 5, lane = tid & 31;
  if (w < B_) {                                            // one wave per batch row
    int b = w;
    float acc = 0.f;
    for (int k = lane; k < KXH; k += 32) {
      float v = (k < H_) ? h[b * H_ + k]
              : (k < 2 * H_) ? ctx[b * H_ + (k - H_)]
              : embf[b * E_ + (k - 2 * H_)];
      acc += v * W_pgen[k];
    }
    for (int m = 16; m; m >>= 1) acc += __shfl_xor(acc, m, 32);
    if (lane == 0) pgen[b] = sigmoidf_(acc + b_pgen[0]);
  }
}

// ---- phase 7: vocab GEMM [16x1024]@[1024x32000] (bf16 WMMA, L2-resident) --
__global__ void k_vocab(const unsigned short* __restrict__ vin16,
                        const unsigned short* __restrict__ wout16,
                        const float* __restrict__ b_out, float* __restrict__ logits) {
  int w = threadIdx.x >> 5, lane = threadIdx.x & 31;
  int half = lane >> 4, n = lane & 15;
  int tile0 = blockIdx.x * 16 + w * 2;                     // 125 blocks x 8 waves x 2 = 2000 tiles
  int n0 = tile0 * 16;
  v8f acc0 = {}, acc1 = {};
  for (int kt = 0; kt < KV / 32; ++kt) {
    Frag a, b0, b1;
    loadA(a, vin16, KV, kt * 32, lane);
    loadB(b0, wout16, KV, n0, kt * 32, lane);
    loadB(b1, wout16, KV, n0 + 16, kt * 32, lane);
    if (kt + 1 < KV / 32)                                  // global_prefetch_b8 of next B tile
      __builtin_prefetch(wout16 + (size_t)(n0 + (lane & 15)) * KV + (kt + 1) * 32, 0, 1);
    acc0 = __builtin_amdgcn_wmma_f32_16x16x32_bf16(false, a.v, false, b0.v,
                                                   (short)0, acc0, false, false);
    acc1 = __builtin_amdgcn_wmma_f32_16x16x32_bf16(false, a.v, false, b1.v,
                                                   (short)0, acc1, false, false);
  }
#pragma unroll
  for (int j = 0; j < 8; ++j) {
    int row = j + 8 * half;
    logits[(size_t)row * V_ + n0 + n]      = acc0[j] + b_out[n0 + n];
    logits[(size_t)row * V_ + n0 + 16 + n] = acc1[j] + b_out[n0 + 16 + n];
  }
}

// ---- phase 8: vocab softmax row statistics --------------------------------
__global__ void k_rowreduce(const float* __restrict__ logits, float* __restrict__ rowmax,
                            float* __restrict__ rowsum) {
  __shared__ float red[256];
  int b = blockIdx.x, tid = threadIdx.x;
  const float* r = logits + (size_t)b * V_;
  float m = -1e30f;
  for (int v = tid; v < V_; v += 256) m = fmaxf(m, r[v]);
  red[tid] = m; __syncthreads();
  for (int o = 128; o; o >>= 1) { if (tid < o) red[tid] = fmaxf(red[tid], red[tid + o]); __syncthreads(); }
  float mx = red[0]; __syncthreads();
  float s = 0.f;
  for (int v = tid; v < V_; v += 256) s += __expf(r[v] - mx);
  red[tid] = s; __syncthreads();
  for (int o = 128; o; o >>= 1) { if (tid < o) red[tid] += red[tid + o]; __syncthreads(); }
  if (tid == 0) { rowmax[b] = mx; rowsum[b] = red[0]; }
}

// ---- phase 9: final = p_gen * softmax padded to EXT -----------------------
__global__ void k_write_gen(int t, const float* __restrict__ logits,
                            const float* __restrict__ rowmax, const float* __restrict__ rowsum,
                            const float* __restrict__ pgen, float* __restrict__ out) {
  int i = blockIdx.x * blockDim.x + threadIdx.x;
  if (i >= B_ * EXT_) return;
  int b = i / EXT_, v = i % EXT_;
  float val = 0.f;
  if (v < V_)
    val = pgen[b] * __expf(logits[(size_t)b * V_ + v] - rowmax[b]) / rowsum[b];
  out[((size_t)b * NSTEP + t) * EXT_ + v] = val;
}

// ---- phase 10: copy-distribution scatter-add ------------------------------
__global__ void k_scatter(int t, const float* __restrict__ attn, const float* __restrict__ pgen,
                          const int* __restrict__ src_ids, const int* __restrict__ src_oov,
                          float* __restrict__ out) {
  int i = blockIdx.x * blockDim.x + threadIdx.x;
  if (i >= B_ * S_) return;
  int b = i / S_;
  int ov = src_oov[i];
  int cid = (ov >= 0) ? (V_ + ov) : src_ids[i];
  atomicAdd(&out[((size_t)b * NSTEP + t) * EXT_ + cid], (1.0f - pgen[b]) * attn[i]);
}

// ---- epilogue: emit hT, cT ------------------------------------------------
__global__ void k_copy_hc(const float* __restrict__ h, const float* __restrict__ c,
                          float* __restrict__ out) {
  int i = blockIdx.x * blockDim.x + threadIdx.x;
  if (i < B_ * H_) {
    size_t base = (size_t)B_ * NSTEP * EXT_;
    out[base + i] = h[i];
    out[base + (size_t)B_ * H_ + i] = c[i];
  }
}

extern "C" void kernel_launch(void* const* d_in, const int* in_sizes, int n_in,
                              void* d_out, int out_size, void* d_ws, size_t ws_size,
                              hipStream_t stream) {
  (void)in_sizes; (void)n_in; (void)out_size; (void)ws_size;
  const int*   tgt       = (const int*)d_in[0];
  const float* hidden    = (const float*)d_in[1];
  const float* cell      = (const float*)d_in[2];
  const float* enc       = (const float*)d_in[3];
  const int*   src_lens  = (const int*)d_in[4];
  const int*   src_ids   = (const int*)d_in[5];
  const int*   src_oov   = (const int*)d_in[6];
  const float* embedding = (const float*)d_in[7];
  const float* W_a       = (const float*)d_in[8];
  const float* W_ih      = (const float*)d_in[9];
  const float* W_hh      = (const float*)d_in[10];
  const float* b_ih      = (const float*)d_in[11];
  const float* b_hh      = (const float*)d_in[12];
  const float* W_out     = (const float*)d_in[13];
  const float* b_out     = (const float*)d_in[14];
  const float* W_pgen    = (const float*)d_in[15];
  const float* b_pgen    = (const float*)d_in[16];
  float* out = (float*)d_out;

  // ---- workspace carve (all 256B aligned) ----
  char* p = (char*)d_ws;
  auto carve = [&](size_t bytes) { char* r = p; p += (bytes + 255) & ~(size_t)255; return r; };
  unsigned short* wout16 = (unsigned short*)carve((size_t)V_ * KV * 2);   // 65.5 MB (L2-resident)
  unsigned short* wcat16 = (unsigned short*)carve((size_t)NG * KXH * 2);  // 5.2 MB
  unsigned short* wa16   = (unsigned short*)carve((size_t)H_ * H_ * 2);
  float* h      = (float*)carve((size_t)B_ * H_ * 4);
  float* c      = (float*)carve((size_t)B_ * H_ * 4);
  float* proj   = (float*)carve((size_t)B_ * H_ * 4);
  float* embf   = (float*)carve((size_t)B_ * E_ * 4);
  float* scores = (float*)carve((size_t)B_ * S_ * 4);
  float* attn   = (float*)carve((size_t)B_ * S_ * 4);
  float* ctx    = (float*)carve((size_t)B_ * H_ * 4);
  unsigned short* h16   = (unsigned short*)carve((size_t)B_ * H_ * 2);
  unsigned short* xh16  = (unsigned short*)carve((size_t)B_ * KXH * 2);
  unsigned short* vin16 = (unsigned short*)carve((size_t)B_ * KV * 2);
  float* gates  = (float*)carve((size_t)B_ * NG * 4);
  float* logits = (float*)carve((size_t)B_ * V_ * 4);                     // 2.0 MB
  float* rowmax = (float*)carve(B_ * 4);
  float* rowsum = (float*)carve(B_ * 4);
  float* pgen   = (float*)carve(B_ * 4);

  // ---- one-time: bf16 weight conversion + state init ----
  k_cvt_wout<<<4096, 256, 0, stream>>>(W_out, wout16);
  k_cvt_wcat<<<1024, 256, 0, stream>>>(W_ih, W_hh, wcat16);
  k_cvt_wa<<<256, 256, 0, stream>>>(W_a, wa16);
  k_init_state<<<(B_ * H_ + 255) / 256, 256, 0, stream>>>(hidden, cell, h, c);

  // ---- 63 serial decoder steps ----
  for (int t = 0; t < NSTEP; ++t) {
    k_proj_embed<<<1, 512, 0, stream>>>(t, tgt, embedding, h, h16, xh16, embf, wa16, proj);
    k_scores<<<dim3(8, 16), 128, 0, stream>>>(proj, enc, src_lens, scores);
    k_softmax<<<16, 256, 0, stream>>>(scores, attn);
    k_context<<<dim3(8, 16), 64, 0, stream>>>(attn, enc, ctx, xh16, vin16);
    k_gates<<<8, 256, 0, stream>>>(xh16, wcat16, b_ih, b_hh, gates);
    k_lstm_pgen<<<1, 512, 0, stream>>>(gates, h, c, ctx, embf, vin16, W_pgen, b_pgen, pgen);
    k_vocab<<<125, 256, 0, stream>>>(vin16, wout16, b_out, logits);
    k_rowreduce<<<16, 256, 0, stream>>>(logits, rowmax, rowsum);
    k_write_gen<<<(B_ * EXT_ + 255) / 256, 256, 0, stream>>>(t, logits, rowmax, rowsum, pgen, out);
    k_scatter<<<(B_ * S_ + 255) / 256, 256, 0, stream>>>(t, attn, pgen, src_ids, src_oov, out);
  }
  k_copy_hc<<<(B_ * H_ + 255) / 256, 256, 0, stream>>>(h, c, out);
}
